// Attention4DDownsample_5669356834801
// MI455X (gfx1250) — matrix operand order repaired and checked by reading, not compiled, not run on hardware
//
#include <hip/hip_runtime.h>

// ---------------------------------------------------------------------------
// Attention4DDownsample for MI455X (gfx1250, wave32, WMMA f16 16x16x32)
// Shapes: B=64 C=384 H=W=28 (N=784), H2=W2=14 (N2=196, padded N2P=208)
//         NHKD=128 (NH=8 heads x KD=16), DH=512 (DV=64/head), OUT=384
// ---------------------------------------------------------------------------

typedef __attribute__((ext_vector_type(16))) _Float16 v16h;
typedef __attribute__((ext_vector_type(8)))  _Float16 v8h;
typedef __attribute__((ext_vector_type(8)))  float    v8f;
typedef __attribute__((ext_vector_type(4)))  float    v4f;
typedef __attribute__((ext_vector_type(4)))  int      v4i;

#define BB    64
#define CC    384
#define HH    28
#define NN    784
#define N2    196
#define N2P   208
#define NH    8
#define KD    16
#define DV    64
#define DHD   512
#define OUTC  384
#define VSTR  800   // padded row stride for v (WMMA K=32 tail reads stay in-bounds)

// ---- gfx1250 async-to-LDS availability (compile-safe fallback) -------------
#if defined(__gfx1250__) && defined(__has_builtin)
#  if __has_builtin(__builtin_amdgcn_global_load_async_to_lds_b128)
#    define HAVE_ASYNC_LDS 1
#  else
#    define HAVE_ASYNC_LDS 0
#  endif
#else
#  define HAVE_ASYNC_LDS 0
#endif

#if HAVE_ASYNC_LDS
#define ASYNC_LDS_B128(gp, lp)                                                  \
    __builtin_amdgcn_global_load_async_to_lds_b128(                             \
        (__attribute__((address_space(1))) v4i*)(void*)(gp),                    \
        (__attribute__((address_space(3))) v4i*)(void*)(lp), 0, 0)
#endif

__device__ __forceinline__ void async_wait0() {
#if defined(__gfx1250__)
#  if defined(__has_builtin) && __has_builtin(__builtin_amdgcn_s_wait_asynccnt)
    __builtin_amdgcn_s_wait_asynccnt(0);
#  else
    asm volatile("s_wait_asynccnt 0" ::: "memory");
#  endif
#endif
}

// ---------------------------------------------------------------------------
// bicubic kernel weight (A = -0.75)
// ---------------------------------------------------------------------------
__device__ __forceinline__ float cubic_w(float x) {
    const float A = -0.75f;
    float ax = fabsf(x);
    if (ax <= 1.0f) return ((A + 2.0f) * ax - (A + 3.0f)) * ax * ax + 1.0f;
    if (ax < 2.0f)  return ((A * ax - 5.0f * A) * ax + 8.0f * A) * ax - 4.0f * A;
    return 0.0f;
}

// ---------------------------------------------------------------------------
// bias[h][p][q] = sum_{a,b} wr[a]*wc[b]*ab_table[h][idxs[ir[a]][ic[b]]]
// ---------------------------------------------------------------------------
__global__ __launch_bounds__(256) void bias_kernel(const float* __restrict__ ab,
                                                   const int* __restrict__ idxs,
                                                   int n_off, float* __restrict__ bias) {
    int idx = blockIdx.x * 256 + threadIdx.x;
    if (idx >= NH * N2 * NN) return;
    int q = idx % NN;
    int p = (idx / NN) % N2;
    int h = idx / (NN * N2);

    float sr = (p + 0.5f) * (16.0f / 196.0f) - 0.5f;
    int   i0 = (int)floorf(sr);
    float tr = sr - (float)i0;
    float sc = (q + 0.5f) * (49.0f / 784.0f) - 0.5f;
    int   j0 = (int)floorf(sc);
    float tc = sc - (float)j0;

    float wr[4], wc[4];
    int   ir[4], ic[4];
#pragma unroll
    for (int o = 0; o < 4; ++o) {
        wr[o] = cubic_w(tr - (float)(o - 1));
        int ii = i0 + o - 1; ir[o] = ii < 0 ? 0 : (ii > 15 ? 15 : ii);
        wc[o] = cubic_w(tc - (float)(o - 1));
        int jj = j0 + o - 1; ic[o] = jj < 0 ? 0 : (jj > 48 ? 48 : jj);
    }
    float acc = 0.0f;
#pragma unroll
    for (int a = 0; a < 4; ++a)
#pragma unroll
        for (int c = 0; c < 4; ++c)
            acc += wr[a] * wc[c] * ab[h * n_off + idxs[ir[a] * 49 + ic[c]]];
    bias[idx] = acc;   // idx == (h*196 + p)*784 + q
}

// ---------------------------------------------------------------------------
// weights f32 -> f16 (one-time; removes cvt chains from GEMM hot loops)
// ---------------------------------------------------------------------------
__global__ __launch_bounds__(256) void wcvt_kernel(const float* __restrict__ src,
                                                   _Float16* __restrict__ dst, int n8) {
    int idx = blockIdx.x * 256 + threadIdx.x;
    if (idx >= n8) return;
    const float* s = src + (size_t)idx * 8;
    v8h o;
#pragma unroll
    for (int u = 0; u < 8; ++u) o[u] = (_Float16)s[u];
    *(v8h*)(dst + (size_t)idx * 8) = o;
}

// ---------------------------------------------------------------------------
// x (B,C,784) f32  ->  xT (B,784,C) f16  (K-contiguous B-fragments)
// ---------------------------------------------------------------------------
__global__ __launch_bounds__(256) void xt_kernel(const float* __restrict__ x,
                                                 _Float16* __restrict__ xT) {
    int idx = blockIdx.x * 256 + threadIdx.x;
    if (idx >= BB * NN * (CC / 8)) return;
    int cg = idx % (CC / 8);
    int n  = (idx / (CC / 8)) % NN;
    int b  = idx / ((CC / 8) * NN);
    const float* xp = x + ((size_t)b * CC + cg * 8) * NN + n;
    v8h o;
#pragma unroll
    for (int u = 0; u < 8; ++u) o[u] = (_Float16)xp[(size_t)u * NN];
    *(v8h*)(xT + ((size_t)b * NN + n) * CC + cg * 8) = o;
}

// ---------------------------------------------------------------------------
// t[b][n2][c] = dwconv3x3s2(x)+q_local_b + pool   (f16, transposed, pad zeroed)
// ---------------------------------------------------------------------------
__global__ __launch_bounds__(256) void t_kernel(const float* __restrict__ x,
                                                const float* __restrict__ qlw,
                                                const float* __restrict__ qlb,
                                                _Float16* __restrict__ t) {
    int idx = blockIdx.x * 256 + threadIdx.x;
    if (idx >= BB * N2P * (CC / 8)) return;
    int cg = idx % (CC / 8);
    int n2 = (idx / (CC / 8)) % N2P;
    int b  = idx / ((CC / 8) * N2P);
    _Float16* dst = t + ((size_t)b * N2P + n2) * CC + cg * 8;
    v8h outv;
    if (n2 >= N2) {
#pragma unroll
        for (int u = 0; u < 8; ++u) outv[u] = (_Float16)0.0f;
        *(v8h*)dst = outv;
        return;
    }
    int i = n2 / 14, j = n2 % 14;
    int ys[3], xs[3];
    bool yv[3], xv[3];
#pragma unroll
    for (int d = 0; d < 3; ++d) {
        int y = 2 * i + d - 1; ys[d] = y; yv[d] = (y >= 0 && y < HH);
        int xx = 2 * j + d - 1; xs[d] = xx; xv[d] = (xx >= 0 && xx < HH);
    }
#pragma unroll
    for (int u = 0; u < 8; ++u) {
        int c = cg * 8 + u;
        const float* xc = x + ((size_t)b * CC + c) * NN;
        const float* wv = qlw + c * 9;
        float acc = qlb[c];
#pragma unroll
        for (int ky = 0; ky < 3; ++ky)
            if (yv[ky])
#pragma unroll
                for (int kx = 0; kx < 3; ++kx)
                    if (xv[kx]) acc += xc[ys[ky] * HH + xs[kx]] * wv[ky * 3 + kx];
        acc += xc[(2 * i) * HH + 2 * j];   // stride-2 "pool" path
        outv[u] = (_Float16)acc;
    }
    *(v8h*)dst = outv;
}

// ---------------------------------------------------------------------------
// v_local: dwconv3x3 s2 over v + bias + BN -> f16 [b][ch][N2P], pad zeroed
// ---------------------------------------------------------------------------
__global__ __launch_bounds__(256) void vloc_kernel(const _Float16* __restrict__ vbuf,
                                                   const float* __restrict__ w,
                                                   const float* __restrict__ lb,
                                                   const float* __restrict__ bs,
                                                   const float* __restrict__ bbv,
                                                   _Float16* __restrict__ vloc) {
    int idx = blockIdx.x * 256 + threadIdx.x;
    if (idx >= BB * DHD * N2P) return;
    int n2 = idx % N2P;
    int ch = (idx / N2P) % DHD;
    int b  = idx / (N2P * DHD);
    _Float16* dst = vloc + (size_t)idx;
    if (n2 >= N2) { *dst = (_Float16)0.0f; return; }
    int i = n2 / 14, j = n2 % 14;
    const _Float16* vp = vbuf + ((size_t)b * DHD + ch) * VSTR;
    const float* wv = w + ch * 9;
    float acc = lb[ch];
#pragma unroll
    for (int ky = 0; ky < 3; ++ky) {
        int y = 2 * i + ky - 1;
        if (y < 0 || y >= HH) continue;
#pragma unroll
        for (int kx = 0; kx < 3; ++kx) {
            int xx = 2 * j + kx - 1;
            if (xx < 0 || xx >= HH) continue;
            acc += (float)vp[y * HH + xx] * wv[ky * 3 + kx];
        }
    }
    *dst = (_Float16)(acc * bs[ch] + bbv[ch]);
}

// ---------------------------------------------------------------------------
// WMMA GEMM, LDS-staged + double-buffered, register-blocked 4x in N:
//   block = 8 waves, block tile = 128(O) x 64(N); wave = 16(O) x 64(N)
//   X (f16, transposed [b][NxRows][C]) tile (64n x 32k = 4KB) is staged to LDS
//   with global_load_async_to_lds_b128 (fallback: load + ds_store), shared by
//   all 8 waves.  Weights pre-converted to f16 (A fragment = 2 x b128).
// Epilogue modes: 0=QT qT[b][h][N2P][KD], 1=KT kT[b][784][128],
//                 2=V v[b][512][VSTR],    3=OUT f32 [b][384][196]
// ---------------------------------------------------------------------------
__global__ __launch_bounds__(256) void gemm_kernel(const _Float16* __restrict__ Wh,
                                                   const float* __restrict__ lb,
                                                   const float* __restrict__ bs,
                                                   const float* __restrict__ bbv,
                                                   const _Float16* __restrict__ X,
                                                   void* __restrict__ Y,
                                                   int C, int NxRows, int Ntiles,
                                                   int obCount, int nbCount, int mode) {
    __shared__ _Float16 Xs[2][64][32];

    int bid = blockIdx.x;
    int nb  = bid % nbCount;
    int ob  = (bid / nbCount) % obCount;
    int b   = bid / (nbCount * obCount);
    int tid = threadIdx.x;
    int lane = tid & 31, wave = tid >> 5;
    int nn = lane & 15, g = lane >> 4;
    int o0 = ob * 128 + wave * 16;
    int nblk = nb * 64;

    // staging: thread -> one 16B segment of the 64x32 f16 tile
    int srow = tid >> 2;            // 0..63
    int sseg = tid & 3;             // 0..3
    int srow_g = nblk + srow;
    if (srow_g > NxRows - 1) srow_g = NxRows - 1;   // clamp pad-tail reads
    const _Float16* sgp = X + ((size_t)b * NxRows + srow_g) * C + sseg * 8;

    const _Float16* Arow = Wh + (size_t)(o0 + nn) * C;

    v8f acc[4] = {};
    int kch = C >> 5;

    // prologue stage of chunk 0
    {
#if HAVE_ASYNC_LDS
        ASYNC_LDS_B128(sgp, &Xs[0][srow][sseg * 8]);
#else
        *(v8h*)&Xs[0][srow][sseg * 8] = *(const v8h*)sgp;
#endif
    }
    async_wait0();
    __syncthreads();

    int cur = 0;
    for (int kc = 0; kc < kch; ++kc) {
        if (kc + 1 < kch) {
            const _Float16* gp = sgp + (kc + 1) * 32;
#if HAVE_ASYNC_LDS
            ASYNC_LDS_B128(gp, &Xs[cur ^ 1][srow][sseg * 8]);
#else
            *(v8h*)&Xs[cur ^ 1][srow][sseg * 8] = *(const v8h*)gp;
#endif
        }
        if (kc + 2 < kch) __builtin_prefetch(Arow + (kc + 2) * 32, 0, 0);

        int c0 = kc * 32;
        // A fragment: elt j<8 -> K=g*8+j ; j>=8 -> K=16+g*8+(j-8)
        v8h a0 = *(const v8h*)(Arow + c0 + g * 8);
        v8h a1 = *(const v8h*)(Arow + c0 + 16 + g * 8);
        v16h a;
#pragma unroll
        for (int i = 0; i < 8; ++i) { a[i] = a0[i]; a[8 + i] = a1[i]; }

#pragma unroll
        for (int t = 0; t < 4; ++t) {
            // B fragment from LDS: elt j<8 -> K = g*8 + j ; j>=8 -> K = 16+g*8+(j-8)
            const _Float16* lp = &Xs[cur][t * 16 + nn][g * 8];
            v8h x0 = *(const v8h*)lp;
            v8h x1 = *(const v8h*)(lp + 16);
            v16h bf;
#pragma unroll
            for (int i = 0; i < 8; ++i) { bf[i] = x0[i]; bf[8 + i] = x1[i]; }
            acc[t] = __builtin_amdgcn_wmma_f32_16x16x32_f16(false, a, false, bf,
                                                            (short)0, acc[t], false, false);
        }
        async_wait0();
        __syncthreads();
        cur ^= 1;
    }

    // epilogue: C/D element (r, lane): M = r + 8*g (=> o), N = lane&15 (=> n)
    int obase = o0 + 8 * g;
    float lbv[8], bsv[8], bbv2[8];
#pragma unroll
    for (int r = 0; r < 8; ++r) {
        int o = obase + r;
        lbv[r] = lb[o]; bsv[r] = bs[o]; bbv2[r] = bbv[o];
    }
#pragma unroll
    for (int t = 0; t < 4; ++t) {
        int ntg = nb * 4 + t;
        if (ntg >= Ntiles) break;
        int n0 = ntg * 16;
        float vals[8];
#pragma unroll
        for (int r = 0; r < 8; ++r) vals[r] = (acc[t][r] + lbv[r]) * bsv[r] + bbv2[r];
        if (mode == 1) {            // kT[b][n][128]
            v8h pk;
#pragma unroll
            for (int r = 0; r < 8; ++r) pk[r] = (_Float16)vals[r];
            *(v8h*)((_Float16*)Y + ((size_t)b * NN + n0 + nn) * 128 + obase) = pk;
        } else if (mode == 0) {     // qT[b][h][N2P][KD]
            v8h pk;
#pragma unroll
            for (int r = 0; r < 8; ++r) pk[r] = (_Float16)vals[r];
            *(v8h*)((_Float16*)Y + (((size_t)b * NH + (o0 >> 4)) * N2P + n0 + nn) * KD + 8 * g) = pk;
        } else if (mode == 2) {     // v[b][ch][VSTR]
#pragma unroll
            for (int r = 0; r < 8; ++r)
                ((_Float16*)Y)[((size_t)b * DHD + obase + r) * VSTR + n0 + nn] = (_Float16)vals[r];
        } else {                    // out f32 [b][384][196]
            int n = n0 + nn;
            if (n < N2) {
#pragma unroll
                for (int r = 0; r < 8; ++r)
                    ((float*)Y)[((size_t)b * OUTC + obase + r) * N2 + n] = vals[r];
            }
        }
    }
}

// ---------------------------------------------------------------------------
// Attention per (b, h, q-tile of 16):
//   QK^T (K=16 zero-padded to 32) * scale + bias -> LDS, softmax, attn@V,
//   + v_local, ReLU, store oT[b][n2][512] f16 for final GEMM.
// ---------------------------------------------------------------------------
__global__ __launch_bounds__(256) void attn_kernel(const _Float16* __restrict__ qT,
                                                   const _Float16* __restrict__ kT,
                                                   const _Float16* __restrict__ vbuf,
                                                   const _Float16* __restrict__ vloc,
                                                   const float* __restrict__ bias,
                                                   _Float16* __restrict__ oT) {
    __shared__ float attn_s[16][800];
    __shared__ float red[16][16];
    __shared__ float rowv[16];
    __shared__ float partC[4][16][16];

    int bid = blockIdx.x;
    int qt = bid % 13;
    int h  = (bid / 13) & 7;
    int b  = bid / (13 * NH);
    int q0 = qt * 16;
    int tid  = threadIdx.x;
    int lane = tid & 31, wave = tid >> 5;
    int nn = lane & 15, g = lane >> 4;

    // zero K-pad columns (784..799) once; 256 threads -> 1 elt each
    attn_s[tid >> 4][784 + (tid & 15)] = 0.0f;

    // A fragment (q): rows M = q-local, K = head dim d (0..15); K 16..31 == 0
    v16h aq = {};
    {
        const _Float16* qp = qT + (((size_t)b * NH + h) * N2P + (q0 + nn)) * KD + g * 8;
        v8h a0 = *(const v8h*)qp;
#pragma unroll
        for (int i = 0; i < 8; ++i) aq[i] = a0[i];
    }

    const float scale = 0.25f;   // KD^-0.5
    for (int nt = wave; nt < 49; nt += 8) {
        int n0 = nt * 16;
        v16h bk = {};
        {                        // B elt j<8 -> K = g*8 + j (d); elts 8..15 are zero pad
            const _Float16* kp = kT + ((size_t)b * NN + n0 + nn) * 128 + h * KD + g * 8;
            v8h k0 = *(const v8h*)kp;
#pragma unroll
            for (int i = 0; i < 8; ++i) { bk[i] = k0[i]; }
        }
        v8f c = {};
        c = __builtin_amdgcn_wmma_f32_16x16x32_f16(false, aq, false, bk,
                                                   (short)0, c, false, false);
#pragma unroll
        for (int r = 0; r < 8; ++r) {
            int M  = r + 8 * g;
            int qq = q0 + M; if (qq > N2 - 1) qq = N2 - 1;     // clamp pad rows
            attn_s[M][n0 + nn] = c[r] * scale + bias[((size_t)h * N2 + qq) * NN + n0 + nn];
        }
    }
    __syncthreads();

    // ---- softmax over 784 cols: 16 rows x 16 threads/row ----
    int row = tid >> 4, col0 = tid & 15;
    float mx = -1e30f;
    for (int cc = col0; cc < NN; cc += 16) mx = fmaxf(mx, attn_s[row][cc]);
    red[row][col0] = mx;
    __syncthreads();
    if (col0 == 0) {
        float mm = red[row][0];
#pragma unroll
        for (int i = 1; i < 16; ++i) mm = fmaxf(mm, red[row][i]);
        rowv[row] = mm;
    }
    __syncthreads();
    float rmax = rowv[row];
    float sum = 0.0f;
    for (int cc = col0; cc < NN; cc += 16) {
        float e = __expf(attn_s[row][cc] - rmax);
        attn_s[row][cc] = e;
        sum += e;
    }
    red[row][col0] = sum;
    __syncthreads();
    if (col0 == 0) {
        float t2 = 0.0f;
#pragma unroll
        for (int i = 0; i < 16; ++i) t2 += red[row][i];
        rowv[row] = 1.0f / t2;
    }
    __syncthreads();

    // ---- attn @ V: 4 dv-tiles x 2 n-halves across 8 waves ----
    int dvt = wave & 3, nhalf = wave >> 2;
    int dv0 = dvt * 16;
    float inv = rowv[nn];        // A-fragment row m = lane&15
    v8f acc = {};
    int ckBeg = nhalf ? 13 : 0;
    int ckEnd = nhalf ? 25 : 13;
    for (int ck = ckBeg; ck < ckEnd; ++ck) {
        int n0 = ck * 32;
        const float* ap0 = &attn_s[nn][n0 + g * 8];
        v4f f0 = *(const v4f*)ap0;
        v4f f1 = *(const v4f*)(ap0 + 4);
        const float* ap1 = &attn_s[nn][n0 + 16 + g * 8];
        v4f f2 = *(const v4f*)ap1;
        v4f f3 = *(const v4f*)(ap1 + 4);
        v16h aa;
#pragma unroll
        for (int i = 0; i < 4; ++i) {
            aa[i]      = (_Float16)(f0[i] * inv);
            aa[4 + i]  = (_Float16)(f1[i] * inv);
            aa[8 + i]  = (_Float16)(f2[i] * inv);
            aa[12 + i] = (_Float16)(f3[i] * inv);
        }
        int ch = h * DV + dv0 + nn;
        const _Float16* vp = vbuf + ((size_t)b * DHD + ch) * VSTR + n0 + g * 8;
        v8h b0 = *(const v8h*)vp;
        v8h b1 = *(const v8h*)(vp + 16);
        v16h bv;
#pragma unroll
        for (int i = 0; i < 8; ++i) { bv[i] = b0[i]; bv[8 + i] = b1[i]; }
        acc = __builtin_amdgcn_wmma_f32_16x16x32_f16(false, aa, false, bv,
                                                     (short)0, acc, false, false);
    }
    if (wave >= 4) {
#pragma unroll
        for (int r = 0; r < 8; ++r) partC[dvt][r + 8 * g][nn] = acc[r];
    }
    __syncthreads();
    if (wave < 4) {
#pragma unroll
        for (int r = 0; r < 8; ++r) acc[r] += partC[dvt][r + 8 * g][nn];
        int ch = h * DV + dv0 + nn;
        const _Float16* vlp = vloc + ((size_t)b * DHD + ch) * N2P + q0 + 8 * g;
        v8h vl = *(const v8h*)vlp;
#pragma unroll
        for (int r = 0; r < 8; ++r) {
            float val = acc[r] + (float)vl[r];
            val = fmaxf(val, 0.0f);                               // relu before final pw
            oT[((size_t)b * N2P + q0 + 8 * g + r) * DHD + ch] = (_Float16)val;
        }
    }
}

// ---------------------------------------------------------------------------
// Host side
// ---------------------------------------------------------------------------
extern "C" void kernel_launch(void* const* d_in, const int* in_sizes, int n_in,
                              void* d_out, int out_size, void* d_ws, size_t ws_size,
                              hipStream_t stream) {
    const float* x    = (const float*)d_in[0];
    const float* qlw  = (const float*)d_in[1];
    const float* qlb  = (const float*)d_in[2];
    const float* qpw  = (const float*)d_in[3];
    const float* qpb  = (const float*)d_in[4];
    const float* qbs  = (const float*)d_in[5];
    const float* qbb  = (const float*)d_in[6];
    const float* kw   = (const float*)d_in[7];
    const float* kb   = (const float*)d_in[8];
    const float* kbs  = (const float*)d_in[9];
    const float* kbb  = (const float*)d_in[10];
    const float* vw   = (const float*)d_in[11];
    const float* vb   = (const float*)d_in[12];
    const float* vbs  = (const float*)d_in[13];
    const float* vbb  = (const float*)d_in[14];
    const float* vlw  = (const float*)d_in[15];
    const float* vlb  = (const float*)d_in[16];
    const float* vlbs = (const float*)d_in[17];
    const float* vlbb = (const float*)d_in[18];
    const float* pw   = (const float*)d_in[19];
    const float* pb   = (const float*)d_in[20];
    const float* pbs  = (const float*)d_in[21];
    const float* pbb  = (const float*)d_in[22];
    const float* ab   = (const float*)d_in[23];
    const int*   bidx = (const int*)d_in[24];
    int n_off = in_sizes[23] / NH;

    char* ws = (char*)d_ws;
    size_t off = 0;
    _Float16* xT   = (_Float16*)(ws + off); off += (size_t)BB * NN * CC * 2;        // 38.5 MB
    _Float16* tbuf = (_Float16*)(ws + off); off += (size_t)BB * N2P * CC * 2;       // 10.2 MB
    _Float16* qTb  = (_Float16*)(ws + off); off += (size_t)BB * NH * N2P * KD * 2;  //  3.4 MB
    _Float16* kTb  = (_Float16*)(ws + off); off += (size_t)BB * NN * 128 * 2;       // 12.8 MB
    _Float16* vbuf = (_Float16*)(ws + off); off += (size_t)BB * DHD * VSTR * 2;     // 52.4 MB
    _Float16* vloc = (_Float16*)(ws + off); off += (size_t)BB * DHD * N2P * 2;      // 13.6 MB
    float*    bias = (float*)(ws + off);    off += (size_t)NH * N2 * NN * 4;        //  4.9 MB
    _Float16* oTb  = (_Float16*)(ws + off); off += (size_t)BB * N2P * DHD * 2;      // 13.6 MB
    _Float16* qpwh = (_Float16*)(ws + off); off += (size_t)128 * CC * 2;
    _Float16* kwh  = (_Float16*)(ws + off); off += (size_t)128 * CC * 2;
    _Float16* vwh  = (_Float16*)(ws + off); off += (size_t)DHD * CC * 2;
    _Float16* pwh  = (_Float16*)(ws + off); off += (size_t)OUTC * DHD * 2;
    (void)ws_size; (void)n_in; (void)out_size;

    float* out = (float*)d_out;

    // 0) weights -> f16
    wcvt_kernel<<<(128 * CC / 8 + 255) / 256, 256, 0, stream>>>(qpw, qpwh, 128 * CC / 8);
    wcvt_kernel<<<(128 * CC / 8 + 255) / 256, 256, 0, stream>>>(kw, kwh, 128 * CC / 8);
    wcvt_kernel<<<(DHD * CC / 8 + 255) / 256, 256, 0, stream>>>(vw, vwh, DHD * CC / 8);
    wcvt_kernel<<<(OUTC * DHD / 8 + 255) / 256, 256, 0, stream>>>(pw, pwh, OUTC * DHD / 8);
    // 1) positional bias table
    bias_kernel<<<(NH * N2 * NN + 255) / 256, 256, 0, stream>>>(ab, bidx, n_off, bias);
    // 2) x -> xT f16
    xt_kernel<<<(BB * NN * (CC / 8) + 255) / 256, 256, 0, stream>>>(x, xT);
    // 3) q input: dwconv + pool, transposed f16
    t_kernel<<<(BB * N2P * (CC / 8) + 255) / 256, 256, 0, stream>>>(x, qlw, qlb, tbuf);
    // 4) q projection GEMM -> qT   (O=128: obCount=1; 13 n-tiles -> 4 blocks)
    gemm_kernel<<<BB * 1 * 4, 256, 0, stream>>>(qpwh, qpb, qbs, qbb, tbuf, qTb,
                                                CC, N2P, 13, 1, 4, 0);
    // 5) k GEMM -> kT              (49 n-tiles -> 13 blocks)
    gemm_kernel<<<BB * 1 * 13, 256, 0, stream>>>(kwh, kb, kbs, kbb, xT, kTb,
                                                 CC, NN, 49, 1, 13, 1);
    // 6) v GEMM -> v               (O=512: obCount=4)
    gemm_kernel<<<BB * 4 * 13, 256, 0, stream>>>(vwh, vb, vbs, vbb, xT, vbuf,
                                                 CC, NN, 49, 4, 13, 2);
    // 7) v_local dwconv
    vloc_kernel<<<(BB * DHD * N2P) / 256, 256, 0, stream>>>(vbuf, vlw, vlb, vlbs, vlbb, vloc);
    // 8) attention (QK^T + bias, softmax, attn@V, +v_local, relu) -> oT
    attn_kernel<<<BB * NH * 13, 256, 0, stream>>>(qTb, kTb, vbuf, vloc, bias, oTb);
    // 9) output projection GEMM -> d_out (f32)  (O=384: obCount=3)
    gemm_kernel<<<BB * 3 * 4, 256, 0, stream>>>(pwh, pb, pbs, pbb, oTb, out,
                                                DHD, N2P, 13, 3, 4, 3);
}
